// AutoregressiveDecoder_88098369176265
// MI455X (gfx1250) — compile-verified, matrix operand order checked by reading, not run on hardware
//
#include <hip/hip_runtime.h>
#include <hip/hip_bf16.h>
#include <math.h>

// ---------------- CDNA5 WMMA types ----------------
typedef __attribute__((ext_vector_type(16))) __bf16 v16bf;
typedef __attribute__((ext_vector_type(8)))  float  v8f;

#define DD    512
#define HH    8
#define HDIM  64
#define LL    6
#define BB    64
#define MCTX  20
#define TT    30
#define SMAX  49   // max kv positions = 20 + 29

// ------------------------------------------------------------------
// Fragment loader: 16 rows x 32 contiguous-K bf16 tile in the CDNA5
// A-matrix layout (lane&15 = row; lane>>4 selects K octet pairing).
// Used for A (X rows) and for B (rows of W == columns of W^T).
// ------------------------------------------------------------------
union FragU { v16bf v; uint4 u[2]; };

__device__ __forceinline__ v16bf load_frag(const __bf16* base, int ld, int row, int k0, int g) {
  const __bf16* r = base + (size_t)row * ld + k0 + g * 8;
  FragU f;
  f.u[0] = *reinterpret_cast<const uint4*>(r);        // K = k0 + g*8 + 0..7
  f.u[1] = *reinterpret_cast<const uint4*>(r + 16);   // K = k0 + 16 + g*8 + 0..7
  return f.v;
}

enum { EPI_F32 = 0, EPI_RES = 1, EPI_GELU_BF = 2, EPI_QKV = 3 };

// ------------------------------------------------------------------
// Generic bf16 WMMA GEMM: Y = epi(A(M,K) @ W(N,K)^T + bias)
// block = 256 threads (8 waves), waves arranged 4(M) x 2(N)
// NT = n-subtiles per wave: block tile = 64M x (NT*32)N
//   NT=4: prefill / precompute (denser reuse per wave)
//   NT=1: decode (more blocks -> more WGPs pulling weights from L2);
//         dual accumulators over interleaved K to break the WMMA chain
// ------------------------------------------------------------------
template<int EPI, int NT>
__global__ __launch_bounds__(256) void gemm_wmma(
    const __bf16* __restrict__ A, int lda,
    const __bf16* __restrict__ W,
    const float*  __restrict__ bias,
    float*  __restrict__ Yf,
    __bf16* __restrict__ Yb, int ldy,
    const float* __restrict__ Res,
    int M, int N, int K,
    float* __restrict__ kc, float* __restrict__ vc,
    int P, int pos_off)
{
  const int w    = threadIdx.x >> 5;
  const int lane = threadIdx.x & 31;
  const int g    = lane >> 4;
  const int lr   = lane & 15;
  const int mbase = blockIdx.y * 64 + (w & 3) * 16;
  const int nbase = blockIdx.x * (NT * 32) + (w >> 2) * (NT * 16);

  v8f acc[NT];
#pragma unroll
  for (int t = 0; t < NT; ++t) {
    v8f z = {0,0,0,0,0,0,0,0};
    acc[t] = z;
  }

  const int arow = mbase + lr;
  const __bf16* wrow = W + (size_t)(nbase + lr) * K;   // prefetch anchor

  if constexpr (NT == 1) {
    // decode: K is a multiple of 64; two independent WMMA chains
    v8f a0 = acc[0], a1 = acc[0];
    for (int k0 = 0; k0 < K; k0 += 64) {
      v16bf x0 = load_frag(A, lda, arow, k0, g);
      v16bf w0 = load_frag(W, K, nbase + lr, k0, g);
      a0 = __builtin_amdgcn_wmma_f32_16x16x32_bf16(false, x0, false, w0, (short)0, a0, false, false);
      v16bf x1 = load_frag(A, lda, arow, k0 + 32, g);
      v16bf w1 = load_frag(W, K, nbase + lr, k0 + 32, g);
      a1 = __builtin_amdgcn_wmma_f32_16x16x32_bf16(false, x1, false, w1, (short)0, a1, false, false);
      __builtin_prefetch(wrow + k0 + 64, 0, 3);   // speculative; dropped if OOB
    }
    acc[0] = a0 + a1;
  } else {
    for (int k0 = 0; k0 < K; k0 += 32) {
      v16bf a = load_frag(A, lda, arow, k0, g);
#pragma unroll
      for (int t = 0; t < NT; ++t) {
        v16bf b = load_frag(W, K, nbase + t * 16 + lr, k0, g);
        acc[t] = __builtin_amdgcn_wmma_f32_16x16x32_bf16(false, a, false, b, (short)0, acc[t], false, false);
      }
      __builtin_prefetch(wrow + k0 + 32, 0, 3);   // speculative; dropped if OOB
    }
  }

#pragma unroll
  for (int t = 0; t < NT; ++t) {
#pragma unroll
    for (int e = 0; e < 8; ++e) {
      const int row = mbase + e + 8 * g;          // D-matrix: M = vgpr + 8*(lane>>4)
      const int col = nbase + t * 16 + lr;        //           N = lane & 15
      float val = acc[t][e] + bias[col];
      if (EPI == EPI_F32) {
        Yf[(size_t)row * ldy + col] = val;
      } else if (EPI == EPI_RES) {
        Yf[(size_t)row * ldy + col] = val + Res[(size_t)row * ldy + col];
      } else if (EPI == EPI_GELU_BF) {
        float gl = 0.5f * val * (1.0f + erff(val * 0.70710678118f));
        Yb[(size_t)row * ldy + col] = (__bf16)gl;
      } else { // EPI_QKV: q -> Yf(M,512); k,v -> per-layer caches at (b,pos)
        const int b   = row / P;
        const int pos = pos_off + (row - b * P);
        if (col < DD)            Yf[(size_t)row * DD + col] = val;
        else if (col < 2 * DD)   kc[((size_t)b * SMAX + pos) * DD + (col - DD)]     = val;
        else                     vc[((size_t)b * SMAX + pos) * DD + (col - 2*DD)]   = val;
      }
    }
  }
}

// ------------------------------------------------------------------
// LayerNorm (f32 in, bf16 out), one wave per row of 512
// ------------------------------------------------------------------
__global__ __launch_bounds__(256) void ln_kernel(
    const float* __restrict__ x, const float* __restrict__ g,
    const float* __restrict__ bt, __bf16* __restrict__ out)
{
  const int w = threadIdx.x >> 5, lane = threadIdx.x & 31;
  const int r = blockIdx.x * 8 + w;
  const float* xr = x + (size_t)r * DD;
  float v[16];
  float s = 0.f;
#pragma unroll
  for (int j = 0; j < 16; ++j) { v[j] = xr[lane + 32 * j]; s += v[j]; }
  for (int m = 16; m; m >>= 1) s += __shfl_xor(s, m);
  const float mean = s * (1.0f / DD);
  float s2 = 0.f;
#pragma unroll
  for (int j = 0; j < 16; ++j) { float d = v[j] - mean; s2 += d * d; }
  for (int m = 16; m; m >>= 1) s2 += __shfl_xor(s2, m);
  const float inv = rsqrtf(s2 * (1.0f / DD) + 1e-5f);
  __bf16* orow = out + (size_t)r * DD;
#pragma unroll
  for (int j = 0; j < 16; ++j) {
    int c = lane + 32 * j;
    orow[c] = (__bf16)((v[j] - mean) * inv * g[c] + bt[c]);
  }
}

// ------------------------------------------------------------------
// Self-attention (causal, KV cache), one wave per (row, head)
// ------------------------------------------------------------------
__global__ __launch_bounds__(256) void self_attn_kernel(
    const float* __restrict__ q, const float* __restrict__ kc,
    const float* __restrict__ vc, __bf16* __restrict__ out,
    int P, int pos_off)
{
  const int h = threadIdx.x >> 5, lane = threadIdx.x & 31;
  const int r = blockIdx.x;
  const int b = r / P;
  const int qpos = pos_off + (r - b * P);
  const int nk = qpos + 1;
  const float* qp = q + (size_t)r * DD + h * HDIM;
  const float qv0 = qp[lane], qv1 = qp[lane + 32];
  const float* kb = kc + (size_t)b * SMAX * DD + h * HDIM;
  const int k0i = lane, k1i = lane + 32;
  const int c0 = k0i < nk ? k0i : nk - 1;   // clamp -> safe reads, uniform shuffles
  const int c1 = k1i < nk ? k1i : nk - 1;
  const float* kr0 = kb + (size_t)c0 * DD;
  const float* kr1 = kb + (size_t)c1 * DD;
  float s0 = 0.f, s1 = 0.f;
  for (int d = 0; d < 32; ++d) { float qd = __shfl(qv0, d); s0 += qd * kr0[d];      s1 += qd * kr1[d]; }
  for (int d = 0; d < 32; ++d) { float qd = __shfl(qv1, d); s0 += qd * kr0[d + 32]; s1 += qd * kr1[d + 32]; }
  s0 = (k0i < nk) ? s0 * 0.125f : -INFINITY;
  s1 = (k1i < nk) ? s1 * 0.125f : -INFINITY;
  float mx = fmaxf(s0, s1);
  for (int m = 16; m; m >>= 1) mx = fmaxf(mx, __shfl_xor(mx, m));
  float p0 = __expf(s0 - mx), p1 = __expf(s1 - mx);
  float sm = p0 + p1;
  for (int m = 16; m; m >>= 1) sm += __shfl_xor(sm, m);
  const float inv = 1.0f / sm; p0 *= inv; p1 *= inv;
  const float* vb = vc + (size_t)b * SMAX * DD + h * HDIM;
  float o0 = 0.f, o1 = 0.f;
  const int n1 = nk < 32 ? nk : 32;
  for (int k = 0; k < n1; ++k) {
    float pk = __shfl(p0, k);
    const float* vr = vb + (size_t)k * DD;
    o0 += pk * vr[lane]; o1 += pk * vr[lane + 32];
  }
  for (int k = 32; k < nk; ++k) {
    float pk = __shfl(p1, k - 32);
    const float* vr = vb + (size_t)k * DD;
    o0 += pk * vr[lane]; o1 += pk * vr[lane + 32];
  }
  out[(size_t)r * DD + h * HDIM + lane]      = (__bf16)o0;
  out[(size_t)r * DD + h * HDIM + lane + 32] = (__bf16)o1;
}

// ------------------------------------------------------------------
// Cross-attention over fixed context (20 keys), kv rows = [K(512)|V(512)]
// ------------------------------------------------------------------
__global__ __launch_bounds__(256) void cross_attn_kernel(
    const float* __restrict__ q, const float* __restrict__ kv,
    __bf16* __restrict__ out, int P)
{
  const int h = threadIdx.x >> 5, lane = threadIdx.x & 31;
  const int r = blockIdx.x;
  const int b = r / P;
  const float* qp = q + (size_t)r * DD + h * HDIM;
  const float qv0 = qp[lane], qv1 = qp[lane + 32];
  const float* kvb = kv + (size_t)b * MCTX * 1024;
  const int kk = lane < MCTX ? lane : MCTX - 1;
  const float* kr = kvb + (size_t)kk * 1024 + h * HDIM;
  float s = 0.f;
  for (int d = 0; d < 32; ++d) s += __shfl(qv0, d) * kr[d];
  for (int d = 0; d < 32; ++d) s += __shfl(qv1, d) * kr[d + 32];
  s = (lane < MCTX) ? s * 0.125f : -INFINITY;
  float mx = s;
  for (int m = 16; m; m >>= 1) mx = fmaxf(mx, __shfl_xor(mx, m));
  float pv = __expf(s - mx);
  float sm = pv;
  for (int m = 16; m; m >>= 1) sm += __shfl_xor(sm, m);
  pv *= (1.0f / sm);
  float o0 = 0.f, o1 = 0.f;
  for (int k = 0; k < MCTX; ++k) {
    float pk = __shfl(pv, k);
    const float* vr = kvb + (size_t)k * 1024 + DD + h * HDIM;
    o0 += pk * vr[lane]; o1 += pk * vr[lane + 32];
  }
  out[(size_t)r * DD + h * HDIM + lane]      = (__bf16)o0;
  out[(size_t)r * DD + h * HDIM + lane + 32] = (__bf16)o1;
}

// ------------------------------------------------------------------
// small utility kernels
// ------------------------------------------------------------------
__global__ void f2bf_kernel(const float* __restrict__ s, __bf16* __restrict__ d, int n) {
  int i = blockIdx.x * 256 + threadIdx.x;
  if (i < n) d[i] = (__bf16)s[i];
}

__global__ void init_prefill_kernel(const float* __restrict__ ctx, const float* __restrict__ pos,
                                    float* __restrict__ x) {
  int i = blockIdx.x * 256 + threadIdx.x;   // i < 1280*512
  int r = i >> 9, c = i & 511;
  int s = r % MCTX;
  x[i] = ctx[i] + pos[s * DD + c];
}

__global__ void extract_last_kernel(const float* __restrict__ xp, float* __restrict__ xd) {
  int i = blockIdx.x * 256 + threadIdx.x;   // i < 64*512
  int b = i >> 9, c = i & 511;
  xd[i] = xp[((size_t)b * MCTX + (MCTX - 1)) * DD + c];
}

__global__ void add_pos_kernel(float* __restrict__ xd, const float* __restrict__ pos, int p) {
  int i = blockIdx.x * 256 + threadIdx.x;   // i < 64*512
  xd[i] += pos[p * DD + (i & 511)];
}

__global__ void logits_kernel(const float* __restrict__ xd, const float* __restrict__ ow,
                              const float* __restrict__ ob, float* __restrict__ out, int t) {
  int tid = threadIdx.x;          // 512 threads: (b, c)
  int b = tid >> 3, c = tid & 7;
  float acc = ob[c];
  const float* xr = xd + (size_t)b * DD;
  const float* wr = ow + (size_t)c * DD;
  for (int k = 0; k < DD; ++k) acc += xr[k] * wr[k];
  out[((size_t)b * TT + t) * 8 + c] = acc;
}

// ------------------------------------------------------------------
// host-side GEMM dispatch: pick tiling by M (decode vs prefill)
// ------------------------------------------------------------------
template<int EPI>
static void launch_gemm(hipStream_t stream,
                        const __bf16* A, int lda, const __bf16* W, const float* bias,
                        float* Yf, __bf16* Yb, int ldy, const float* Res,
                        int M, int N, int K,
                        float* kc, float* vc, int P, int pos_off) {
  if (M <= 64) {
    gemm_wmma<EPI, 1><<<dim3(N / 32, M / 64), 256, 0, stream>>>(
        A, lda, W, bias, Yf, Yb, ldy, Res, M, N, K, kc, vc, P, pos_off);
  } else {
    gemm_wmma<EPI, 4><<<dim3(N / 128, M / 64), 256, 0, stream>>>(
        A, lda, W, bias, Yf, Yb, ldy, Res, M, N, K, kc, vc, P, pos_off);
  }
}

// ------------------------------------------------------------------
// host launch
// ------------------------------------------------------------------
extern "C" void kernel_launch(void* const* d_in, const int* in_sizes, int n_in,
                              void* d_out, int out_size, void* d_ws, size_t ws_size,
                              hipStream_t stream) {
  const float* ctx   = (const float*)d_in[0];
  const float* pos   = (const float*)d_in[1];
  const float* sa_w  = (const float*)d_in[2];
  const float* sa_b  = (const float*)d_in[3];
  const float* sa_ow = (const float*)d_in[4];
  const float* sa_ob = (const float*)d_in[5];
  const float* ca_w  = (const float*)d_in[6];
  const float* ca_b  = (const float*)d_in[7];
  const float* ca_ow = (const float*)d_in[8];
  const float* ca_ob = (const float*)d_in[9];
  const float* ln1g  = (const float*)d_in[10];
  const float* ln1b  = (const float*)d_in[11];
  const float* ln2g  = (const float*)d_in[12];
  const float* ln2b  = (const float*)d_in[13];
  const float* ln3g  = (const float*)d_in[14];
  const float* ln3b  = (const float*)d_in[15];
  const float* f1w   = (const float*)d_in[16];
  const float* f1b   = (const float*)d_in[17];
  const float* f2w   = (const float*)d_in[18];
  const float* f2b   = (const float*)d_in[19];
  const float* outw  = (const float*)d_in[20];
  const float* outb  = (const float*)d_in[21];
  float* dout = (float*)d_out;

  // ---- carve workspace ----
  char* wp = (char*)d_ws;
  auto carve = [&](size_t bytes) -> void* {
    void* r = (void*)wp;
    wp += (bytes + 255) & ~(size_t)255;
    return r;
  };
  __bf16* w_sa   = (__bf16*)carve((size_t)LL * 1536 * 512 * 2);
  __bf16* w_saow = (__bf16*)carve((size_t)LL * 512 * 512 * 2);
  __bf16* w_ca   = (__bf16*)carve((size_t)LL * 1536 * 512 * 2);
  __bf16* w_caow = (__bf16*)carve((size_t)LL * 512 * 512 * 2);
  __bf16* w_f1   = (__bf16*)carve((size_t)LL * 2048 * 512 * 2);
  __bf16* w_f2   = (__bf16*)carve((size_t)LL * 512 * 2048 * 2);
  __bf16* ctx_bf = (__bf16*)carve((size_t)BB * MCTX * DD * 2);
  float*  ca_kv  = (float*)carve((size_t)LL * BB * MCTX * 1024 * 4);
  float*  kcache = (float*)carve((size_t)LL * BB * SMAX * DD * 4);
  float*  vcache = (float*)carve((size_t)LL * BB * SMAX * DD * 4);
  float*  x_pre  = (float*)carve((size_t)BB * MCTX * DD * 4);
  float*  x_dec  = (float*)carve((size_t)BB * DD * 4);
  float*  qbuf   = (float*)carve((size_t)BB * MCTX * DD * 4);
  __bf16* xa     = (__bf16*)carve((size_t)BB * MCTX * DD * 2);
  __bf16* abf    = (__bf16*)carve((size_t)BB * MCTX * DD * 2);
  __bf16* hbuf   = (__bf16*)carve((size_t)BB * MCTX * 2048 * 2);

  // ---- weight conversion fp32 -> bf16 ----
  auto cvt = [&](const float* s, __bf16* d, int n) {
    f2bf_kernel<<<(n + 255) / 256, 256, 0, stream>>>(s, d, n);
  };
  cvt(sa_w,  w_sa,   LL * 1536 * 512);
  cvt(sa_ow, w_saow, LL * 512 * 512);
  cvt(ca_w,  w_ca,   LL * 1536 * 512);
  cvt(ca_ow, w_caow, LL * 512 * 512);
  cvt(f1w,   w_f1,   LL * 2048 * 512);
  cvt(f2w,   w_f2,   LL * 512 * 2048);
  cvt(ctx,   ctx_bf, BB * MCTX * DD);

  // ---- precompute cross-attn K/V per layer: ctx @ ca_w[i][512:1536]^T + b ----
  for (int i = 0; i < LL; ++i) {
    launch_gemm<EPI_F32>(stream,
        ctx_bf, DD,
        w_ca + (size_t)i * 1536 * 512 + (size_t)512 * 512,
        ca_b + (size_t)i * 1536 + 512,
        ca_kv + (size_t)i * BB * MCTX * 1024, nullptr, 1024,
        nullptr, BB * MCTX, 1024, DD, nullptr, nullptr, 1, 0);
  }

  // ---- per-chunk transformer stack ----
  auto run_chunk = [&](float* x, int M, int P, int pos_off) {
    for (int i = 0; i < LL; ++i) {
      const __bf16* Wsa   = w_sa   + (size_t)i * 1536 * 512;
      const __bf16* Wsaow = w_saow + (size_t)i * 512 * 512;
      const __bf16* Wcaq  = w_ca   + (size_t)i * 1536 * 512;
      const __bf16* Wcaow = w_caow + (size_t)i * 512 * 512;
      const __bf16* Wf1   = w_f1   + (size_t)i * 2048 * 512;
      const __bf16* Wf2   = w_f2   + (size_t)i * 512 * 2048;
      float* kc_i = kcache + (size_t)i * BB * SMAX * DD;
      float* vc_i = vcache + (size_t)i * BB * SMAX * DD;
      const float* kv_i = ca_kv + (size_t)i * BB * MCTX * 1024;

      // --- self-attention sublayer ---
      ln_kernel<<<M / 8, 256, 0, stream>>>(x, ln1g + i * DD, ln1b + i * DD, xa);
      launch_gemm<EPI_QKV>(stream,
          xa, DD, Wsa, sa_b + (size_t)i * 1536, qbuf, nullptr, DD,
          nullptr, M, 1536, DD, kc_i, vc_i, P, pos_off);
      self_attn_kernel<<<M, 256, 0, stream>>>(qbuf, kc_i, vc_i, abf, P, pos_off);
      launch_gemm<EPI_RES>(stream,
          abf, DD, Wsaow, sa_ob + (size_t)i * DD, x, nullptr, DD,
          x, M, 512, DD, nullptr, nullptr, 1, 0);

      // --- cross-attention sublayer ---
      ln_kernel<<<M / 8, 256, 0, stream>>>(x, ln2g + i * DD, ln2b + i * DD, xa);
      launch_gemm<EPI_F32>(stream,
          xa, DD, Wcaq, ca_b + (size_t)i * 1536, qbuf, nullptr, DD,
          nullptr, M, 512, DD, nullptr, nullptr, 1, 0);
      cross_attn_kernel<<<M, 256, 0, stream>>>(qbuf, kv_i, abf, P);
      launch_gemm<EPI_RES>(stream,
          abf, DD, Wcaow, ca_ob + (size_t)i * DD, x, nullptr, DD,
          x, M, 512, DD, nullptr, nullptr, 1, 0);

      // --- FFN sublayer ---
      ln_kernel<<<M / 8, 256, 0, stream>>>(x, ln3g + i * DD, ln3b + i * DD, xa);
      launch_gemm<EPI_GELU_BF>(stream,
          xa, DD, Wf1, f1b + (size_t)i * 2048, nullptr, hbuf, 2048,
          nullptr, M, 2048, DD, nullptr, nullptr, 1, 0);
      launch_gemm<EPI_RES>(stream,
          hbuf, 2048, Wf2, f2b + (size_t)i * DD, x, nullptr, DD,
          x, M, 512, 2048, nullptr, nullptr, 1, 0);
    }
  };

  // ---- prefill: S = 20, positions 0..19 ----
  {
    const int n = BB * MCTX * DD;
    init_prefill_kernel<<<n / 256, 256, 0, stream>>>(ctx, pos, x_pre);
  }
  run_chunk(x_pre, BB * MCTX, MCTX, 0);
  extract_last_kernel<<<(BB * DD) / 256, 256, 0, stream>>>(x_pre, x_dec);
  logits_kernel<<<1, 512, 0, stream>>>(x_dec, outw, outb, dout, 0);
  add_pos_kernel<<<(BB * DD) / 256, 256, 0, stream>>>(x_dec, pos, MCTX);

  // ---- decode steps t = 1..29 (new position p = 19 + t) ----
  for (int t = 1; t < TT; ++t) {
    const int p = MCTX - 1 + t;
    run_chunk(x_dec, BB, 1, p);
    logits_kernel<<<1, 512, 0, stream>>>(x_dec, outw, outb, dout, t);
    if (t < TT - 1)
      add_pos_kernel<<<(BB * DD) / 256, 256, 0, stream>>>(x_dec, pos, p + 1);
  }
}
